// ReEig_43319040148098
// MI455X (gfx1250) — matrix operand order repaired and checked by reading, not compiled, not run on hardware
//
#include <hip/hip_runtime.h>

typedef __attribute__((ext_vector_type(16))) _Float16 v16h;
typedef __attribute__((ext_vector_type(8)))  _Float16 v8h;
typedef __attribute__((ext_vector_type(4)))  _Float16 v4h;
typedef __attribute__((ext_vector_type(8)))  float    v8f;

#define DIM   64
#define NITER 18
#define EPS   1e-4f

// ---- WMMA: D = A(16x32 f16) * B(32x16 f16) + C(16x16 f32) ----
__device__ __forceinline__ v8f wmma_mac(v16h a, v16h b, v8f c) {
  return __builtin_amdgcn_wmma_f32_16x16x32_f16(
      false, a, false, b, (short)0, c, false, false);
}

// A fragment (16x32 f16) from row-major S (ISA 7.12.2):
// lanes 0-15:  row=16ti+lane,    K = {0..7,16..23}+32ks
// lanes 16-31: row=16ti+lane-16, K = {8..15,24..31}+32ks
__device__ __forceinline__ v16h load_a_frag(const _Float16* S, int ti, int ks, int lane) {
  const int row = 16 * ti + (lane & 15);
  const int k0  = 32 * ks + ((lane & 16) ? 8 : 0);
  const v8h* p  = (const v8h*)(S + row * DIM + k0);
  union { v16h v; v8h h[2]; } u;
  u.h[0] = p[0];
  u.h[1] = p[2];          // +16 halves
  return u.v;
}

// B fragment (32x16 f16): column n of the operand == row n of the SAME buffer
// (all operands here are symmetric), elements 0..15 = K = {k0..k0+7, k0+16..k0+23}.
__device__ __forceinline__ v16h load_b_frag(const _Float16* S, int tj, int ks, int lane) {
  const int col = 16 * tj + (lane & 15);
  const int k0  = 32 * ks + ((lane & 16) ? 8 : 0);
  const v8h* p  = (const v8h*)(S + col * DIM + k0);
  union { v16h v; v8h h[2]; } u;
  u.h[0] = p[0];
  u.h[1] = p[2];
  return u.v;
}

// C/D tile: VGPR r -> row m0+r, col fixed per lane. Since the result is
// symmetric, storing it "transposed" (contiguous 16B per tile) IS the matrix.
__device__ __forceinline__ v8h tile_to_h(v8f c) {
  v8h h;
#pragma unroll
  for (int r = 0; r < 8; ++r) h[r] = (_Float16)c[r];
  return h;
}

__device__ __forceinline__ void store_tile(_Float16* S, int ti, int tj, int lane, v8h h) {
  const int col = 16 * tj + (lane & 15);
  const int m0  = 16 * ti + ((lane & 16) ? 8 : 0);
  *(v8h*)(S + col * DIM + m0) = h;
}

__global__ __launch_bounds__(128) void reeig_ns_kernel(const float* __restrict__ X,
                                                       float* __restrict__ Out) {
  __shared__ __align__(16) float    Xs[DIM * DIM];       // 16 KB original (f32)
  __shared__ __align__(16) _Float16 Aq[DIM * DIM];       //  8 KB (X-epsI)/c
  __shared__ __align__(16) _Float16 Ybuf[2][DIM * DIM];  // 16 KB NS iterate
  __shared__ __align__(16) _Float16 Zs[DIM * DIM];       //  8 KB Y*Y
  __shared__ float red[4];
  __shared__ float cshare;

  const int tid  = threadIdx.x;
  const int wave = tid >> 5;
  const int lane = tid & 31;
  const float* Xg = X + (size_t)blockIdx.x * DIM * DIM;
  float*       Og = Out + (size_t)blockIdx.x * DIM * DIM;

  // ---- load X (b128 global loads) ----
  {
    const float4* src = (const float4*)Xg;
    float4*       dst = (float4*)Xs;
#pragma unroll 2
    for (int i = tid; i < DIM * DIM / 4; i += 128) dst[i] = src[i];
  }
  __syncthreads();

  // ---- Frobenius norm of M = X - eps I (>= ||M||_2 -> NS converges) ----
  float acc = 0.f;
  for (int i4 = tid; i4 < DIM * DIM / 4; i4 += 128) {
    const int r = (i4 * 4) >> 6, c0 = (i4 * 4) & 63;
    float4 x = ((const float4*)Xs)[i4];
    if (r == c0 + 0) x.x -= EPS;
    if (r == c0 + 1) x.y -= EPS;
    if (r == c0 + 2) x.z -= EPS;
    if (r == c0 + 3) x.w -= EPS;
    acc += x.x * x.x + x.y * x.y + x.z * x.z + x.w * x.w;
  }
#pragma unroll
  for (int off = 16; off; off >>= 1) acc += __shfl_xor(acc, off, 32);
  if (lane == 0) red[wave] = acc;
  __syncthreads();
  if (tid == 0) cshare = sqrtf(red[0] + red[1] + red[2] + red[3]) + 1e-20f;
  __syncthreads();
  const float cnorm = cshare;
  const float inv_c = 1.0f / cnorm;

  // ---- Aq = Y0 = (X - eps I)/c in f16 (symmetric -> one copy serves A and B) ----
  for (int i4 = tid; i4 < DIM * DIM / 4; i4 += 128) {
    const int r = (i4 * 4) >> 6, c0 = (i4 * 4) & 63;
    float4 x = ((const float4*)Xs)[i4];
    if (r == c0 + 0) x.x -= EPS;
    if (r == c0 + 1) x.y -= EPS;
    if (r == c0 + 2) x.z -= EPS;
    if (r == c0 + 3) x.w -= EPS;
    v4h h;
    h[0] = (_Float16)(x.x * inv_c);
    h[1] = (_Float16)(x.y * inv_c);
    h[2] = (_Float16)(x.z * inv_c);
    h[3] = (_Float16)(x.w * inv_c);
    ((v4h*)Aq)[i4]      = h;
    ((v4h*)Ybuf[0])[i4] = h;
  }
  __syncthreads();

  // ---- Newton–Schulz sign iteration: Y <- 1.5*Y - 0.5*Y^3 ----
  int cur = 0;
  for (int it = 0; it < NITER; ++it) {
    const _Float16* Yn = &Ybuf[cur][0];
    // Z = Y*Y (symmetric; stored once, tile-contiguous)
    {
      v8f c0 = {}, c1 = {}, c2 = {}, c3 = {};
#pragma unroll
      for (int ks = 0; ks < 2; ++ks) {
        const v16h a = load_a_frag(Yn, wave, ks, lane);
        c0 = wmma_mac(a, load_b_frag(Yn, 0, ks, lane), c0);
        c1 = wmma_mac(a, load_b_frag(Yn, 1, ks, lane), c1);
        c2 = wmma_mac(a, load_b_frag(Yn, 2, ks, lane), c2);
        c3 = wmma_mac(a, load_b_frag(Yn, 3, ks, lane), c3);
      }
      store_tile(Zs, wave, 0, lane, tile_to_h(c0));
      store_tile(Zs, wave, 1, lane, tile_to_h(c1));
      store_tile(Zs, wave, 2, lane, tile_to_h(c2));
      store_tile(Zs, wave, 3, lane, tile_to_h(c3));
    }
    __syncthreads();
    // Ynew = 1.5*Y - 0.5*(Y*Z)
    {
      v8f c0 = {}, c1 = {}, c2 = {}, c3 = {};
#pragma unroll
      for (int ks = 0; ks < 2; ++ks) {
        const v16h a = load_a_frag(Yn, wave, ks, lane);
        c0 = wmma_mac(a, load_b_frag(Zs, 0, ks, lane), c0);
        c1 = wmma_mac(a, load_b_frag(Zs, 1, ks, lane), c1);
        c2 = wmma_mac(a, load_b_frag(Zs, 2, ks, lane), c2);
        c3 = wmma_mac(a, load_b_frag(Zs, 3, ks, lane), c3);
      }
      const int m0 = 16 * wave + ((lane & 16) ? 8 : 0);
      const int cn = lane & 15;
      v8f cc[4] = {c0, c1, c2, c3};
#pragma unroll
      for (int tj = 0; tj < 4; ++tj) {
        const int col = 16 * tj + cn;
        const v8h y = *(const v8h*)(Yn + col * DIM + m0);  // contiguous (symmetry)
        v8h h;
#pragma unroll
        for (int r = 0; r < 8; ++r)
          h[r] = (_Float16)(1.5f * (float)y[r] - 0.5f * cc[tj][r]);
        store_tile(&Ybuf[cur ^ 1][0], wave, tj, lane, h);
      }
    }
    __syncthreads();
    cur ^= 1;
  }

  // ---- |M| = c*(A'*sign);  Out = 0.5*(X + |M|) + 0.5*eps*I  (symmetric) ----
  {
    const _Float16* Sg = &Ybuf[cur][0];
    v8f c0 = {}, c1 = {}, c2 = {}, c3 = {};
#pragma unroll
    for (int ks = 0; ks < 2; ++ks) {
      const v16h a = load_a_frag(Aq, wave, ks, lane);
      c0 = wmma_mac(a, load_b_frag(Sg, 0, ks, lane), c0);
      c1 = wmma_mac(a, load_b_frag(Sg, 1, ks, lane), c1);
      c2 = wmma_mac(a, load_b_frag(Sg, 2, ks, lane), c2);
      c3 = wmma_mac(a, load_b_frag(Sg, 3, ks, lane), c3);
    }
    const int m0 = 16 * wave + ((lane & 16) ? 8 : 0);
    const int cn = lane & 15;
    v8f cc[4] = {c0, c1, c2, c3};
#pragma unroll
    for (int tj = 0; tj < 4; ++tj) {
      const int col = 16 * tj + cn;
      // Out symmetric -> write the lane's row-run at (col, m0..m0+7):
      // contiguous 32B per lane = wide global stores; X read is contiguous too.
      const float4 x0 = *(const float4*)(Xs + col * DIM + m0);
      const float4 x1 = *(const float4*)(Xs + col * DIM + m0 + 4);
      float v[8];
#pragma unroll
      for (int r = 0; r < 8; ++r) {
        const float xv = (r < 4) ? (&x0.x)[r] : (&x1.x)[r - 4];
        v[r] = 0.5f * xv + 0.5f * cnorm * cc[tj][r];
        if (m0 + r == col) v[r] += 0.5f * EPS;
      }
      *(float4*)(Og + col * DIM + m0)     = make_float4(v[0], v[1], v[2], v[3]);
      *(float4*)(Og + col * DIM + m0 + 4) = make_float4(v[4], v[5], v[6], v[7]);
    }
  }
}

extern "C" void kernel_launch(void* const* d_in, const int* in_sizes, int n_in,
                              void* d_out, int out_size, void* d_ws, size_t ws_size,
                              hipStream_t stream) {
  const float* X   = (const float*)d_in[0];
  float*       Out = (float*)d_out;
  const int B = in_sizes[0] / (DIM * DIM);
  reeig_ns_kernel<<<dim3(B), dim3(128), 0, stream>>>(X, Out);
}
